// AttentionModule3D_62225486184841
// MI455X (gfx1250) — compile-verified, matrix operand order checked by reading, not run on hardware
//
#include <hip/hip_runtime.h>

// ---------------------------------------------------------------------------
// AttentionModule3D for MI455X (gfx1250, wave32)
// x:(2,64,64,64,64) f32. BN -> dw-conv5 -> dilated dw-conv7 -> 1x1x1 conv
// (WMMA bf16) -> gate with x.
// ---------------------------------------------------------------------------

typedef __attribute__((ext_vector_type(16))) __bf16 v16bf;
typedef __attribute__((ext_vector_type(8)))  float  v8f;

#define NB 2            // batch
#define CH 64           // channels
#define SP 64           // spatial dim
#define VOX (SP*SP*SP)  // 262144 voxels per (n,c)
#define CNT (NB*VOX)    // 524288 elems per channel (BN reduction size)

// ---------------------------------------------------------------------------
// BatchNorm statistics: deterministic two-pass reduction.
// grid (64 slices, 64 channels), 256 threads. Each block reduces 8192 elems.
// ---------------------------------------------------------------------------
__global__ __launch_bounds__(256)
void bn_stats_kernel(const float* __restrict__ x,
                     float* __restrict__ part_sum,   // [CH][64]
                     float* __restrict__ part_sq) {  // [CH][64]
  __shared__ float ss[256];
  __shared__ float sq[256];
  const int c   = blockIdx.y;
  const int tid = threadIdx.x;
  float s = 0.f, q = 0.f;
  size_t j = (size_t)blockIdx.x * 8192 + tid;
#pragma unroll 4
  for (int i = 0; i < 32; ++i, j += 256) {
    const size_t n = j >> 18;            // j / VOX
    const size_t r = j & (VOX - 1);
    const float v = x[(((n * CH) + c) << 18) + r];
    s += v; q += v * v;
  }
  ss[tid] = s; sq[tid] = q;
  __syncthreads();
  for (int off = 128; off > 0; off >>= 1) {
    if (tid < off) { ss[tid] += ss[tid + off]; sq[tid] += sq[tid + off]; }
    __syncthreads();
  }
  if (tid == 0) {
    part_sum[c * 64 + blockIdx.x] = ss[0];
    part_sq [c * 64 + blockIdx.x] = sq[0];
  }
}

__global__ __launch_bounds__(64)
void bn_finalize_kernel(const float* __restrict__ part_sum,
                        const float* __restrict__ part_sq,
                        const float* __restrict__ gamma,
                        const float* __restrict__ beta,
                        float* __restrict__ scale,
                        float* __restrict__ shift) {
  const int c = threadIdx.x;
  float s = 0.f, q = 0.f;
  for (int i = 0; i < 64; ++i) { s += part_sum[c * 64 + i]; q += part_sq[c * 64 + i]; }
  const float inv = 1.f / (float)CNT;
  const float m   = s * inv;
  const float var = q * inv - m * m;
  const float sc  = gamma[c] * rsqrtf(var + 1e-5f);
  scale[c] = sc;
  shift[c] = beta[c] - m * sc;
}

// ---------------------------------------------------------------------------
// Depthwise conv 5^3, pad 2, BN affine fused into the tile load.
// block (8,8,8); grid (512 tiles, 64 ch, 2 batch). 12^3 f32 LDS tile (~6.9KB).
// ---------------------------------------------------------------------------
__global__ __launch_bounds__(512)
void dw5_kernel(const float* __restrict__ x,
                const float* __restrict__ w0,     // [64][125]
                const float* __restrict__ bias0,  // [64]
                const float* __restrict__ scale,
                const float* __restrict__ shift,
                __bf16* __restrict__ attn0) {
  __shared__ float s_in[12 * 12 * 12];
  __shared__ float s_w[125];
  const int t  = blockIdx.x;
  const int c  = blockIdx.y;
  const int n  = blockIdx.z;
  const int bd = (t >> 6) * 8, bh = ((t >> 3) & 7) * 8, bw = (t & 7) * 8;
  const float sc = scale[c], sh = shift[c];
  const float* __restrict__ xc = x + (((size_t)n * CH + c) << 18);
  const int tid = threadIdx.x + 8 * threadIdx.y + 64 * threadIdx.z;

  for (int i = tid; i < 12 * 12 * 12; i += 512) {
    const int lw = i % 12, lh = (i / 12) % 12, ld = i / 144;
    const int gd = bd + ld - 2, gh = bh + lh - 2, gw = bw + lw - 2;
    float v = 0.f;
    if ((unsigned)gd < SP && (unsigned)gh < SP && (unsigned)gw < SP)
      v = xc[((gd * SP) + gh) * SP + gw] * sc + sh;
    s_in[i] = v;
  }
  if (tid < 125) s_w[tid] = w0[c * 125 + tid];
  __syncthreads();

  const int lw = threadIdx.x, lh = threadIdx.y, ld = threadIdx.z;
  float acc = bias0[c];
#pragma unroll
  for (int kd = 0; kd < 5; ++kd)
#pragma unroll
    for (int kh = 0; kh < 5; ++kh)
#pragma unroll
      for (int kw = 0; kw < 5; ++kw)
        acc += s_w[(kd * 5 + kh) * 5 + kw] *
               s_in[((ld + kd) * 12 + (lh + kh)) * 12 + (lw + kw)];

  attn0[(((size_t)n * CH + c) << 18) +
        (((bd + ld) * SP) + (bh + lh)) * SP + (bw + lw)] = (__bf16)acc;
}

// ---------------------------------------------------------------------------
// Depthwise conv 7^3, dilation 3, pad 9.
// Dynamic LDS: 26^3 f32 halo tile + 343 f32 weights (~71.7 KB).
// bf16 -> f32 conversion happens once at staging, not per-tap.
// ---------------------------------------------------------------------------
__global__ __launch_bounds__(512)
void dw7_kernel(const __bf16* __restrict__ attn0,
                const float* __restrict__ wsw,    // [64][343]
                const float* __restrict__ biass,  // [64]
                __bf16* __restrict__ attn1) {
  extern __shared__ float dsmem[];
  float* s_in = dsmem;            // 26*26*26 = 17576 floats
  float* s_w  = dsmem + 17576;    // 343 floats
  const int t  = blockIdx.x;
  const int c  = blockIdx.y;
  const int n  = blockIdx.z;
  const int bd = (t >> 6) * 8, bh = ((t >> 3) & 7) * 8, bw = (t & 7) * 8;
  const __bf16* __restrict__ ac = attn0 + (((size_t)n * CH + c) << 18);
  const int tid = threadIdx.x + 8 * threadIdx.y + 64 * threadIdx.z;

  for (int i = tid; i < 26 * 26 * 26; i += 512) {
    const int lw = i % 26, lh = (i / 26) % 26, ld = i / 676;
    const int gd = bd + ld - 9, gh = bh + lh - 9, gw = bw + lw - 9;
    float v = 0.f;
    if ((unsigned)gd < SP && (unsigned)gh < SP && (unsigned)gw < SP)
      v = (float)ac[((gd * SP) + gh) * SP + gw];
    s_in[i] = v;
  }
  if (tid < 343) s_w[tid] = wsw[c * 343 + tid];
  __syncthreads();

  const int lw = threadIdx.x, lh = threadIdx.y, ld = threadIdx.z;
  float acc = biass[c];
  for (int kd = 0; kd < 7; ++kd) {
#pragma unroll
    for (int kh = 0; kh < 7; ++kh)
#pragma unroll
      for (int kw = 0; kw < 7; ++kw)
        acc += s_w[(kd * 7 + kh) * 7 + kw] *
               s_in[((ld + 3 * kd) * 26 + (lh + 3 * kh)) * 26 + (lw + 3 * kw)];
  }
  attn1[(((size_t)n * CH + c) << 18) +
        (((bd + ld) * SP) + (bh + lh)) * SP + (bw + lw)] = (__bf16)acc;
}

// ---------------------------------------------------------------------------
// Repack W1 (64x64 f32) into WMMA bf16 A-fragment layout:
// wpk[((t*2+kc)*32 + lane)*16 + i], 16-bit A layout per CDNA5 ISA 7.12.2:
//   lanes 0-15  -> K = {0..7, 16..23}, lanes 16-31 -> K = {8..15, 24..31}
// ---------------------------------------------------------------------------
__global__ __launch_bounds__(256)
void prepack_w1_kernel(const float* __restrict__ w1, __bf16* __restrict__ wpk) {
  const int tid   = threadIdx.x;          // 256 = 4 tiles * 2 kchunks * 32 lanes
  const int lane  = tid & 31;
  const int kc    = (tid >> 5) & 1;
  const int t     = tid >> 6;
  const int m     = lane & 15;
  const int khalf = lane >> 4;
#pragma unroll
  for (int i = 0; i < 16; ++i) {
    const int v = i >> 1, p = i & 1;
    const int k = kc * 32 + khalf * 8 + (v < 4 ? 2 * v + p : 16 + 2 * (v - 4) + p);
    wpk[tid * 16 + i] = (__bf16)w1[(t * 16 + m) * CH + k];
  }
}

// ---------------------------------------------------------------------------
// Pointwise 1x1x1 conv (64x64 channel matmul) via v_wmma_f32_16x16x32_bf16,
// fused with bias add and residual gate out = x * attn.
// Block = 256 threads (8 waves), each wave owns 16 voxels, all 64 c_out.
// attn tile is staged into LDS *pre-swizzled into B-fragment order* so each
// lane's 32-byte fragment is one contiguous ds_load_b128 pair (no u16 gather).
// LDS layout: s_frag[((waveTile*2 + kChunk)*32 + lane)*16 + fi]
//   inverse lane map for element k (0..63):  kc = k>>5, kr = k&31,
//   khalf = (kr>>3)&1, fi = (kr&7) | ((kr&16)>>1), lane = khalf*16 + ncol.
// ---------------------------------------------------------------------------
__global__ __launch_bounds__(256)
void pw_wmma_gate_kernel(const float*  __restrict__ x,
                         const __bf16* __restrict__ wpk,
                         const float*  __restrict__ b1,
                         const __bf16* __restrict__ attn1,
                         float* __restrict__ out) {
  __shared__ __bf16 s_frag[8 * 2 * 32 * 16];  // 8192 bf16 = 16 KB
  const int tid    = threadIdx.x;
  const int blk    = blockIdx.x;              // 4096 = 2 * (262144/128)
  const int n      = blk >> 11;
  const int vblock = (blk & 2047) * 128;
  const __bf16* __restrict__ A1 = attn1 + ((size_t)n * CH << 18);

  // Coalesced global read (8B per thread-iter), swizzled scatter into LDS.
  for (int i = tid * 4; i < CH * 128; i += 256 * 4) {
    const int row = i >> 7;   // k index (0..63)
    const int col = i & 127;  // voxel base within tile (multiple of 4)
    const uint2 d = *(const uint2*)&A1[((size_t)row << 18) + vblock + col];
    const __bf16* ev = (const __bf16*)&d;
    const int kc    = row >> 5;
    const int kr    = row & 31;
    const int khalf = (kr >> 3) & 1;
    const int fi    = (kr & 7) | ((kr & 16) >> 1);
#pragma unroll
    for (int j = 0; j < 4; ++j) {
      const int v     = col + j;
      const int wt    = v >> 4;           // wave tile 0..7
      const int lane_ = khalf * 16 + (v & 15);
      s_frag[(((wt * 2 + kc) * 32) + lane_) * 16 + fi] = ev[j];
    }
  }
  __syncthreads();

  const int lane  = tid & 31;
  const int wave  = tid >> 5;
  const int ncol  = lane & 15;
  const int khalf = lane >> 4;

  // Whole 32B fragments: each is a contiguous aligned LDS load.
  v16bf bfrag[2];
  bfrag[0] = *(const v16bf*)&s_frag[(((wave * 2 + 0) * 32) + lane) * 16];
  bfrag[1] = *(const v16bf*)&s_frag[(((wave * 2 + 1) * 32) + lane) * 16];

  const int voxel = vblock + wave * 16 + ncol;
#pragma unroll
  for (int t = 0; t < 4; ++t) {
    v8f acc = {};
#pragma unroll
    for (int kc = 0; kc < 2; ++kc) {
      const v16bf afrag = *(const v16bf*)&wpk[(((t * 2 + kc) * 32) + lane) * 16];
      acc = __builtin_amdgcn_wmma_f32_16x16x32_bf16(
          false, afrag, false, bfrag[kc], (short)0, acc, false, false);
    }
    // C/D layout: VGPR r -> M = r (+8 for lanes 16-31), N = lane%16.
#pragma unroll
    for (int r = 0; r < 8; ++r) {
      const int cout = t * 16 + r + khalf * 8;
      const size_t idx = (((size_t)n * CH + cout) << 18) + voxel;
      out[idx] = x[idx] * (acc[r] + b1[cout]);
    }
  }
}

// ---------------------------------------------------------------------------
extern "C" void kernel_launch(void* const* d_in, const int* in_sizes, int n_in,
                              void* d_out, int out_size, void* d_ws, size_t ws_size,
                              hipStream_t stream) {
  const float* x     = (const float*)d_in[0];
  const float* gamma = (const float*)d_in[1];
  const float* beta  = (const float*)d_in[2];
  const float* w0    = (const float*)d_in[3];
  const float* b0    = (const float*)d_in[4];
  const float* wsw   = (const float*)d_in[5];
  const float* bs    = (const float*)d_in[6];
  const float* w1    = (const float*)d_in[7];
  const float* b1    = (const float*)d_in[8];
  float* out = (float*)d_out;

  // Workspace layout: partials(32KB) | scale/shift | wpk bf16 | attn0 | attn1
  float*  ws_psum  = (float*)d_ws;                   // 64*64
  float*  ws_psq   = ws_psum + 64 * 64;              // 64*64
  float*  ws_scale = ws_psq + 64 * 64;               // 64
  float*  ws_shift = ws_scale + 64;                  // 64
  __bf16* ws_wpk   = (__bf16*)(ws_shift + 64);       // 4096 bf16
  __bf16* ws_attn0 = ws_wpk + 4096;                  // 33.5M bf16 (64 MiB)
  __bf16* ws_attn1 = ws_attn0 + (size_t)NB * CH * VOX;

  bn_stats_kernel<<<dim3(64, 64), 256, 0, stream>>>(x, ws_psum, ws_psq);
  bn_finalize_kernel<<<1, 64, 0, stream>>>(ws_psum, ws_psq, gamma, beta,
                                           ws_scale, ws_shift);
  prepack_w1_kernel<<<1, 256, 0, stream>>>(w1, ws_wpk);
  dw5_kernel<<<dim3(512, CH, NB), dim3(8, 8, 8), 0, stream>>>(
      x, w0, b0, ws_scale, ws_shift, ws_attn0);
  const size_t dw7_lds = (size_t)(26 * 26 * 26 + 343) * sizeof(float);
  dw7_kernel<<<dim3(512, CH, NB), dim3(8, 8, 8), dw7_lds, stream>>>(
      ws_attn0, wsw, bs, ws_attn1);
  pw_wmma_gate_kernel<<<4096, 256, 0, stream>>>(x, ws_wpk, b1, ws_attn1, out);
}
